// ReinforceSender_78512002171224
// MI455X (gfx1250) — matrix operand order repaired and checked by reading, not compiled, not run on hardware
//
#include <hip/hip_runtime.h>
#include <hip/hip_bf16.h>
#include <stdint.h>

// ---------------------------------------------------------------------------
// Fused REINFORCE sender (encoder + 6-step GRU decoder) for gfx1250.
// All GEMMs via v_wmma_f32_16x16x32_bf16; all recurrent state lives in LDS.
// Each wave32 owns 16 batch rows for the whole sequence -> no barriers.
// ---------------------------------------------------------------------------

typedef __bf16 bf16_t;
typedef __bf16 v16bf __attribute__((ext_vector_type(16)));
typedef float  v8f  __attribute__((ext_vector_type(8)));

#define B_TOT  131072
#define D_IN   128
#define HDIM   256
#define VTOT   6
#define TSTEPS 6
#define WAVES  4
#define BT     (WAVES * 16)   // batch rows per block

__device__ __forceinline__ float sigmoidf_(float x) { return 1.f / (1.f + __expf(-x)); }

// counter-based hash -> uniform -> gumbel (deterministic stand-in for threefry)
__device__ __forceinline__ uint32_t mix32(uint32_t a, uint32_t b, uint32_t c) {
  uint32_t h = a * 0x9E3779B9u ^ b * 0x85EBCA6Bu ^ c * 0xC2B2AE35u;
  h ^= h >> 16; h *= 0x7FEB352Du; h ^= h >> 15; h *= 0x846CA68Bu; h ^= h >> 16;
  return h;
}

// A-fragment (16x32 bf16, MxK): lane<16 holds K {k0..k0+15},
// lane>=16 holds K {k0+16..k0+31}; row m = lane&15. Two 16B LDS loads.
__device__ __forceinline__ v16bf load_a_frag(const bf16_t* base, int ldk, int m, int k0, int akb) {
  union { v16bf v; struct { uint4 lo; uint4 hi; } q; } f;
  const bf16_t* p = base + m * ldk + k0 + akb;
  f.q.lo = *(const uint4*)(p);
  f.q.hi = *(const uint4*)(p + 8);
  return f.v;
}

__global__ __launch_bounds__(WAVES * 32)
void sender_kernel(const float* __restrict__ x,
                   const bf16_t* __restrict__ WtEnc,   // [H][D_IN]   (n-major)
                   const bf16_t* __restrict__ Wtz,     // [H][2H]
                   const bf16_t* __restrict__ Wtr,     // [H][2H]
                   const bf16_t* __restrict__ Wth,     // [H][2H]
                   const bf16_t* __restrict__ WtOut,   // [16][H] (cols 6..15 zero)
                   const float* __restrict__ b_enc,
                   const float* __restrict__ b_z,
                   const float* __restrict__ b_r,
                   const float* __restrict__ b_h,
                   const float* __restrict__ b_out,
                   const float* __restrict__ W_embed,  // [VTOT][H]
                   const float* __restrict__ b_embed,
                   const float* __restrict__ start_embed,
                   float* __restrict__ msg,            // [B][T*VTOT]
                   float* __restrict__ out_lp,         // [B]
                   int*   __restrict__ out_len)        // [B]
{
  __shared__ bf16_t hS  [WAVES][16][HDIM];
  __shared__ bf16_t xinS[WAVES][16][HDIM];
  __shared__ bf16_t zS  [WAVES][16][HDIM];
  __shared__ bf16_t rhS [WAVES][16][HDIM];   // r*h (also reused to stage x as bf16)
  __shared__ float  lbuf[WAVES][16][16];
  __shared__ int    tokS[WAVES][16];
  __shared__ float  actS[WAVES][16];

  const int tid  = threadIdx.x;
  const int w    = tid >> 5;
  const int l    = tid & 31;
  const int nl   = l & 15;                  // tile column / A row
  const int akb  = (l < 16) ? 0 : 16;       // A K sub-offset
  const int bkb  = (l < 16) ? 0 : 16;       // B K sub-offset
  const int crow = (l < 16) ? 0 : 8;        // C row base
  const int rowBase = blockIdx.x * BT + w * 16;

  bf16_t* hP   = &hS[w][0][0];
  bf16_t* xinP = &xinS[w][0][0];
  bf16_t* rhP  = &rhS[w][0][0];

  // ---- stage x rows as bf16 (reuse rh buffer), init xin = start_embed ----
  const float* xr = x + (size_t)rowBase * D_IN;
  for (int i = l; i < 16 * D_IN; i += 32) rhP[i] = (bf16_t)xr[i];
  for (int i = l; i < 16 * HDIM; i += 32) xinP[i] = (bf16_t)start_embed[i & (HDIM - 1)];

  // ---- encoder: h0 = x @ W_enc + b_enc ----
  for (int nt = 0; nt < 16; ++nt) {
    v8f acc = {0.f, 0.f, 0.f, 0.f, 0.f, 0.f, 0.f, 0.f};
    const bf16_t* wc = WtEnc + (size_t)(nt * 16 + nl) * D_IN + bkb;
    for (int kt = 0; kt < D_IN / 32; ++kt) {
      v16bf a = load_a_frag(rhP, D_IN, nl, kt * 32, akb);
      v16bf b = *(const v16bf*)(wc + kt * 32);
      acc = __builtin_amdgcn_wmma_f32_16x16x32_bf16(false, a, false, b, (short)0, acc, false, false);
    }
    const int col = nt * 16 + nl;
    const float bb = b_enc[col];
    for (int v = 0; v < 8; ++v) hS[w][crow + v][col] = (bf16_t)(acc[v] + bb);
  }

  float tlp = 0.f; int lenr = 0; bool stopped = false;

  for (int t = 0; t < TSTEPS; ++t) {
    // ---- z = sigmoid([xin,h] @ W_z + b_z) ----
    for (int nt = 0; nt < 16; ++nt) {
      v8f acc = {0.f, 0.f, 0.f, 0.f, 0.f, 0.f, 0.f, 0.f};
      const bf16_t* wc = Wtz + (size_t)(nt * 16 + nl) * (2 * HDIM) + bkb;
      for (int kt = 0; kt < 8; ++kt) {
        v16bf a = load_a_frag(xinP, HDIM, nl, kt * 32, akb);
        v16bf b = *(const v16bf*)(wc + kt * 32);
        acc = __builtin_amdgcn_wmma_f32_16x16x32_bf16(false, a, false, b, (short)0, acc, false, false);
      }
      for (int kt = 0; kt < 8; ++kt) {
        v16bf a = load_a_frag(hP, HDIM, nl, kt * 32, akb);
        v16bf b = *(const v16bf*)(wc + HDIM + kt * 32);
        acc = __builtin_amdgcn_wmma_f32_16x16x32_bf16(false, a, false, b, (short)0, acc, false, false);
      }
      const int col = nt * 16 + nl;
      const float bb = b_z[col];
      for (int v = 0; v < 8; ++v) zS[w][crow + v][col] = (bf16_t)sigmoidf_(acc[v] + bb);
    }
    // ---- r = sigmoid([xin,h] @ W_r + b_r); rh = r*h ----
    for (int nt = 0; nt < 16; ++nt) {
      v8f acc = {0.f, 0.f, 0.f, 0.f, 0.f, 0.f, 0.f, 0.f};
      const bf16_t* wc = Wtr + (size_t)(nt * 16 + nl) * (2 * HDIM) + bkb;
      for (int kt = 0; kt < 8; ++kt) {
        v16bf a = load_a_frag(xinP, HDIM, nl, kt * 32, akb);
        v16bf b = *(const v16bf*)(wc + kt * 32);
        acc = __builtin_amdgcn_wmma_f32_16x16x32_bf16(false, a, false, b, (short)0, acc, false, false);
      }
      for (int kt = 0; kt < 8; ++kt) {
        v16bf a = load_a_frag(hP, HDIM, nl, kt * 32, akb);
        v16bf b = *(const v16bf*)(wc + HDIM + kt * 32);
        acc = __builtin_amdgcn_wmma_f32_16x16x32_bf16(false, a, false, b, (short)0, acc, false, false);
      }
      const int col = nt * 16 + nl;
      const float bb = b_r[col];
      for (int v = 0; v < 8; ++v) {
        float r = sigmoidf_(acc[v] + bb);
        rhS[w][crow + v][col] = (bf16_t)(r * (float)hS[w][crow + v][col]);
      }
    }
    // ---- h_t = tanh([xin, r*h] @ W_h + b_h); h = (1-z)*h + z*h_t ----
    for (int nt = 0; nt < 16; ++nt) {
      v8f acc = {0.f, 0.f, 0.f, 0.f, 0.f, 0.f, 0.f, 0.f};
      const bf16_t* wc = Wth + (size_t)(nt * 16 + nl) * (2 * HDIM) + bkb;
      for (int kt = 0; kt < 8; ++kt) {
        v16bf a = load_a_frag(xinP, HDIM, nl, kt * 32, akb);
        v16bf b = *(const v16bf*)(wc + kt * 32);
        acc = __builtin_amdgcn_wmma_f32_16x16x32_bf16(false, a, false, b, (short)0, acc, false, false);
      }
      for (int kt = 0; kt < 8; ++kt) {
        v16bf a = load_a_frag(rhP, HDIM, nl, kt * 32, akb);
        v16bf b = *(const v16bf*)(wc + HDIM + kt * 32);
        acc = __builtin_amdgcn_wmma_f32_16x16x32_bf16(false, a, false, b, (short)0, acc, false, false);
      }
      const int col = nt * 16 + nl;
      const float bb = b_h[col];
      for (int v = 0; v < 8; ++v) {
        float ht = tanhf(acc[v] + bb);
        float zz = (float)zS[w][crow + v][col];
        float ho = (float)hS[w][crow + v][col];
        hS[w][crow + v][col] = (bf16_t)((1.f - zz) * ho + zz * ht);
      }
    }
    // ---- logits = h @ W_out (padded to 16 cols) ----
    {
      v8f acc = {0.f, 0.f, 0.f, 0.f, 0.f, 0.f, 0.f, 0.f};
      const bf16_t* wc = WtOut + (size_t)nl * HDIM + bkb;
      for (int kt = 0; kt < 8; ++kt) {
        v16bf a = load_a_frag(hP, HDIM, nl, kt * 32, akb);
        v16bf b = *(const v16bf*)(wc + kt * 32);
        acc = __builtin_amdgcn_wmma_f32_16x16x32_bf16(false, a, false, b, (short)0, acc, false, false);
      }
      for (int v = 0; v < 8; ++v) lbuf[w][crow + v][nl] = acc[v];
    }
    // ---- sampling + log-softmax bookkeeping (one row per lane, lanes 0..15) ----
    if (l < 16) {
      const int r = l;
      const int brow = rowBase + r;
      float lg[VTOT];
      for (int j = 0; j < VTOT; ++j) lg[j] = lbuf[w][r][j] + b_out[j];
      int tok = 0; float best = -3.0e38f;
      for (int j = 0; j < VTOT; ++j) {
        uint32_t hsh = mix32(42u + (uint32_t)t, (uint32_t)brow, (uint32_t)j);
        float u = fminf(fmaxf((float)hsh * 2.3283064365386963e-10f, 1e-9f), 1.f - 1e-7f);
        float g = -__logf(-__logf(u));
        float val = lg[j] + g;
        if (val > best) { best = val; tok = j; }
      }
      float mx = lg[0];
      for (int j = 1; j < VTOT; ++j) mx = fmaxf(mx, lg[j]);
      float s = 0.f;
      for (int j = 0; j < VTOT; ++j) s += __expf(lg[j] - mx);
      float lse = mx + __logf(s);
      float active = stopped ? 0.f : 1.f;
      tlp += (lg[tok] - lse) * active;
      float* mrow = msg + (size_t)brow * (TSTEPS * VTOT) + t * VTOT;
      for (int j = 0; j < VTOT; ++j) mrow[j] = (j == tok) ? active : 0.f;
      bool is_stop = (tok == VTOT - 1);
      if (!stopped && !is_stop) ++lenr;
      tokS[w][r] = tok; actS[w][r] = active;
      stopped = stopped || is_stop;
    }
    // ---- prev_embed = active * W_embed[token] + b_embed (all 32 lanes) ----
    for (int r = 0; r < 16; ++r) {
      const int tok = tokS[w][r];
      const float act = actS[w][r];
      const float* we = W_embed + (size_t)tok * HDIM;
      for (int j = 0; j < 8; ++j) {
        const int c = l * 8 + j;
        xinS[w][r][c] = (bf16_t)(act * we[c] + b_embed[c]);
      }
    }
  }

  if (l < 16) {
    const int brow = rowBase + l;
    out_lp[brow] = tlp;
    out_len[brow] = lenr;
  }
}

// ---- weight preprocessing: transpose (n-major) + f32 -> bf16 ----
__global__ void transpose_bf16_kernel(const float* __restrict__ W, bf16_t* __restrict__ Wt,
                                      int K, int N) {
  int idx = blockIdx.x * blockDim.x + threadIdx.x;
  if (idx >= K * N) return;
  int n = idx / K, k = idx - n * K;
  Wt[idx] = (bf16_t)W[(size_t)k * N + n];
}

__global__ void transpose_out_kernel(const float* __restrict__ W, bf16_t* __restrict__ Wt) {
  // W: [256][6] -> Wt: [16][256], rows 6..15 zero-padded
  int idx = blockIdx.x * blockDim.x + threadIdx.x;
  if (idx >= 16 * HDIM) return;
  int n = idx / HDIM, k = idx - n * HDIM;
  Wt[idx] = (bf16_t)((n < VTOT) ? W[k * VTOT + n] : 0.f);
}

extern "C" void kernel_launch(void* const* d_in, const int* in_sizes, int n_in,
                              void* d_out, int out_size, void* d_ws, size_t ws_size,
                              hipStream_t stream) {
  const float* x       = (const float*)d_in[0];
  const float* W_enc   = (const float*)d_in[1];
  const float* b_enc   = (const float*)d_in[2];
  const float* W_embed = (const float*)d_in[3];
  const float* b_embed = (const float*)d_in[4];
  const float* W_z     = (const float*)d_in[5];
  const float* b_z     = (const float*)d_in[6];
  const float* W_r     = (const float*)d_in[7];
  const float* b_r     = (const float*)d_in[8];
  const float* W_h     = (const float*)d_in[9];
  const float* b_h     = (const float*)d_in[10];
  const float* W_out   = (const float*)d_in[11];
  const float* b_out   = (const float*)d_in[12];
  const float* start   = (const float*)d_in[13];

  char* ws = (char*)d_ws;
  bf16_t* WtEnc = (bf16_t*)(ws);                          // 256*128*2 = 64KB
  bf16_t* Wtz   = (bf16_t*)(ws + 65536);                  // 256*512*2 = 256KB
  bf16_t* Wtr   = (bf16_t*)(ws + 65536 + 262144);
  bf16_t* Wth   = (bf16_t*)(ws + 65536 + 2 * 262144);
  bf16_t* WtOut = (bf16_t*)(ws + 65536 + 3 * 262144);     // 16*256*2 = 8KB

  transpose_bf16_kernel<<<(HDIM * D_IN + 255) / 256, 256, 0, stream>>>(W_enc, WtEnc, D_IN, HDIM);
  transpose_bf16_kernel<<<(HDIM * 2 * HDIM + 255) / 256, 256, 0, stream>>>(W_z, Wtz, 2 * HDIM, HDIM);
  transpose_bf16_kernel<<<(HDIM * 2 * HDIM + 255) / 256, 256, 0, stream>>>(W_r, Wtr, 2 * HDIM, HDIM);
  transpose_bf16_kernel<<<(HDIM * 2 * HDIM + 255) / 256, 256, 0, stream>>>(W_h, Wth, 2 * HDIM, HDIM);
  transpose_out_kernel<<<(16 * HDIM + 255) / 256, 256, 0, stream>>>(W_out, WtOut);

  float* msg    = (float*)d_out;
  float* out_lp = msg + (size_t)B_TOT * (TSTEPS * VTOT);
  int*   out_len = (int*)(msg + (size_t)B_TOT * (TSTEPS * VTOT + 1));

  sender_kernel<<<B_TOT / BT, WAVES * 32, 0, stream>>>(
      x, WtEnc, Wtz, Wtr, Wth, WtOut,
      b_enc, b_z, b_r, b_h, b_out, W_embed, b_embed, start,
      msg, out_lp, out_len);
}
